// ScaledDotProductAttention_14568529068116
// MI455X (gfx1250) — compile-verified, matrix operand order checked by reading, not run on hardware
//
#include <hip/hip_runtime.h>
#include <cmath>

// CDNA5 WMMA fragment types (wave32)
typedef __attribute__((ext_vector_type(16))) __bf16 v16bf;
typedef __attribute__((ext_vector_type(8)))  float  v8f;

// Problem constants (reference: B,H,S,D = 2,16,2048,128)
constexpr int   Bc = 2, Hc = 16, Sc = 2048, Dc = 128;
constexpr float SCALE   = 0.08838834764831845f;   // 1/sqrt(128)
constexpr float NEG_BIG = -1.0e9f;

constexpr int BQ            = 16;                 // query rows per workgroup
constexpr int WAVES         = 8;                  // 256 threads
constexpr int KEYS_PER_WAVE = Sc / WAVES;         // 256
constexpr int TILES         = KEYS_PER_WAVE / 16; // 16 score tiles per wave

// 16-bit A/B fragment element -> K offset within a 16x32 (or 32x16) tile.
// From ISA 7.12.2: VGPR v holds K = {0,1},{2,3},... lanes>=16 add +8; v>=4 adds +16.
__device__ __forceinline__ int koffset(int e, int hi) {
    const int v = e >> 1;
    return ((v & 3) * 2) + ((v >> 2) * 16) + hi * 8 + (e & 1);
}

__global__ __launch_bounds__(256)
void attn_fused_kernel(const float* __restrict__ Q, const float* __restrict__ K,
                       const float* __restrict__ V, const float* __restrict__ Mask,
                       float* __restrict__ Out, float* __restrict__ W)
{
    extern __shared__ char lds[];                       // WAVES*8KB reuse region + redbuf
    float* redbuf = (float*)(lds + WAVES * 8192);       // [WAVES][16] reduction scratch

    const int tid  = threadIdx.x;
    const int wave = tid >> 5;
    const int lane = tid & 31;
    const int lm   = lane & 15;   // N (column) / M (row) selector within fragment
    const int hi   = lane >> 4;   // half-wave selector

    const int    bh     = blockIdx.y;       // b*H + h, 0..31
    const int    b      = bh >> 4;          // H == 16
    const int    qbase  = blockIdx.x * BQ;
    const size_t rowBase = (size_t)bh * Sc; // row index base in (bh, S) space

    // ---------------- Phase 0: Q tile -> A fragments (scale folded in) ----------------
    v16bf qa[4];
    {
        const float* qrow = Q + (rowBase + qbase + lm) * Dc;   // this lane's query row
        #pragma unroll
        for (int c = 0; c < 4; ++c)
            #pragma unroll
            for (int e = 0; e < 16; ++e)
                qa[c][e] = (__bf16)(qrow[c * 32 + koffset(e, hi)] * SCALE);
    }

    // ---------------- Phase 1: scores S = (Q*scale) K^T + mask*NEG_BIG ----------------
    const int wkey0 = wave * KEYS_PER_WAVE;
    v8f acc[TILES];
    #pragma unroll
    for (int t = 0; t < TILES; ++t) {
        const int    key  = wkey0 + t * 16 + lm;               // this lane's key column
        const float* krow = K + (rowBase + key) * Dc;
        v16bf kb[4];
        #pragma unroll
        for (int c = 0; c < 4; ++c)
            #pragma unroll
            for (int e = 0; e < 16; ++e)
                kb[c][e] = (__bf16)krow[c * 32 + koffset(e, hi)];
        v8f cfrag = {0.f, 0.f, 0.f, 0.f, 0.f, 0.f, 0.f, 0.f};
        #pragma unroll
        for (int cc = 0; cc < 4; ++cc)
            cfrag = __builtin_amdgcn_wmma_f32_16x16x32_bf16(
                        false, qa[cc], false, kb[cc], (short)0, cfrag, false, false);
        const float mv = Mask[(size_t)b * Sc + key] * NEG_BIG; // per-column mask
        #pragma unroll
        for (int r = 0; r < 8; ++r) cfrag[r] += mv;
        acc[t] = cfrag;
    }

    // ---------------- Phase 2: row max (lane -> half-wave -> workgroup) ----------------
    float rmax[8];
    #pragma unroll
    for (int r = 0; r < 8; ++r) rmax[r] = -3.0e38f;
    #pragma unroll
    for (int t = 0; t < TILES; ++t)
        #pragma unroll
        for (int r = 0; r < 8; ++r) rmax[r] = fmaxf(rmax[r], acc[t][r]);
    #pragma unroll
    for (int off = 8; off; off >>= 1)
        #pragma unroll
        for (int r = 0; r < 8; ++r)
            rmax[r] = fmaxf(rmax[r], __shfl_xor(rmax[r], off, 32));
    if (lm == 0)
        #pragma unroll
        for (int r = 0; r < 8; ++r) redbuf[wave * 16 + hi * 8 + r] = rmax[r];
    __syncthreads();
    #pragma unroll
    for (int r = 0; r < 8; ++r) {
        float m = -3.0e38f;
        for (int w2 = 0; w2 < WAVES; ++w2) m = fmaxf(m, redbuf[w2 * 16 + hi * 8 + r]);
        rmax[r] = m;
    }
    __syncthreads();

    // ---------------- Phase 3: exp + row sum ----------------
    float rsum[8];
    #pragma unroll
    for (int r = 0; r < 8; ++r) rsum[r] = 0.f;
    #pragma unroll
    for (int t = 0; t < TILES; ++t)
        #pragma unroll
        for (int r = 0; r < 8; ++r) {
            const float ev = __expf(acc[t][r] - rmax[r]);
            acc[t][r] = ev;
            rsum[r]  += ev;
        }
    #pragma unroll
    for (int off = 8; off; off >>= 1)
        #pragma unroll
        for (int r = 0; r < 8; ++r) rsum[r] += __shfl_xor(rsum[r], off, 32);
    if (lm == 0)
        #pragma unroll
        for (int r = 0; r < 8; ++r) redbuf[wave * 16 + hi * 8 + r] = rsum[r];
    __syncthreads();
    #pragma unroll
    for (int r = 0; r < 8; ++r) {
        float s = 0.f;
        for (int w2 = 0; w2 < WAVES; ++w2) s += redbuf[w2 * 16 + hi * 8 + r];
        rsum[r] = 1.0f / s;
    }

    // ---------------- Phase 4: write attention weights + stage P (bf16) in LDS --------
    __bf16* Pl = (__bf16*)(lds + wave * 8192);   // [16 rows][256 keys] per-wave slice
    #pragma unroll
    for (int t = 0; t < TILES; ++t) {
        const int col = wkey0 + t * 16 + lm;
        #pragma unroll
        for (int r = 0; r < 8; ++r) {
            const int   row = hi * 8 + r;
            const float p   = acc[t][r] * rsum[r];
            W[(rowBase + qbase + row) * (size_t)Sc + col] = p;        // 537MB stream-out
            Pl[row * KEYS_PER_WAVE + t * 16 + lm] = (__bf16)p;       // restage for PV
        }
    }

    // ---------------- Phase 5: O_partial = P(16x256) * V(256x128) via WMMA ------------
    v8f oacc[8];
    #pragma unroll
    for (int nt = 0; nt < 8; ++nt) oacc[nt] = (v8f){0.f,0.f,0.f,0.f,0.f,0.f,0.f,0.f};
    #pragma unroll
    for (int kc = 0; kc < 8; ++kc) {                 // 8 chunks of 32 keys
        v16bf a;
        #pragma unroll
        for (int e = 0; e < 16; ++e)
            a[e] = Pl[lm * KEYS_PER_WAVE + kc * 32 + koffset(e, hi)];
        #pragma unroll
        for (int nt = 0; nt < 8; ++nt) {             // 8 tiles of 16 output dims
            v16bf bfrag;
            #pragma unroll
            for (int e = 0; e < 16; ++e) {
                const int key = wkey0 + kc * 32 + koffset(e, hi);
                bfrag[e] = (__bf16)V[(rowBase + key) * Dc + nt * 16 + lm];
            }
            oacc[nt] = __builtin_amdgcn_wmma_f32_16x16x32_bf16(
                           false, a, false, bfrag, (short)0, oacc[nt], false, false);
        }
    }

    // ---------------- Phase 6: cross-wave O reduction through LDS ---------------------
    float* Ol = (float*)(lds + wave * 8192);         // reuse wave's own slice (DS in-order)
    #pragma unroll
    for (int nt = 0; nt < 8; ++nt)
        #pragma unroll
        for (int r = 0; r < 8; ++r)
            Ol[(hi * 8 + r) * Dc + nt * 16 + lm] = oacc[nt][r];
    __syncthreads();

    #pragma unroll
    for (int i = 0; i < 8; ++i) {                    // 2048 outputs, 8 per thread
        const int e = tid + i * 256;
        float s = 0.f;
        for (int w2 = 0; w2 < WAVES; ++w2) s += ((const float*)lds)[w2 * 2048 + e];
        const int row = e >> 7, d = e & 127;
        Out[(rowBase + qbase + row) * Dc + d] = s;
    }
}

extern "C" void kernel_launch(void* const* d_in, const int* in_sizes, int n_in,
                              void* d_out, int out_size, void* d_ws, size_t ws_size,
                              hipStream_t stream) {
    (void)in_sizes; (void)n_in; (void)out_size; (void)d_ws; (void)ws_size;
    const float* Q = (const float*)d_in[0];
    const float* K = (const float*)d_in[1];
    const float* V = (const float*)d_in[2];
    const float* M = (const float*)d_in[3];
    float* Out = (float*)d_out;                                   // [B,H,S,D] first
    float* W   = Out + (size_t)Bc * Hc * Sc * Dc;                 // then [B,H,S,S]

    dim3 grid(Sc / BQ, Bc * Hc);                                  // (128, 32)
    const size_t ldsBytes = WAVES * 8192 + WAVES * 16 * sizeof(float); // 66048 B
    hipLaunchKernelGGL(attn_fused_kernel, grid, dim3(256), ldsBytes, stream,
                       Q, K, V, M, Out, W);
}
